// GCN_Encoder_33148557591177
// MI455X (gfx1250) — matrix operand order repaired and checked by reading, not compiled, not run on hardware
//
#include <hip/hip_runtime.h>

typedef __attribute__((ext_vector_type(16))) _Float16 v16h;
typedef __attribute__((ext_vector_type(8)))  float    v8f;

#define TPB 256
#define WPB 8   // waves per block (wave32)

static __device__ __forceinline__ v16h zv16() {
  v16h r;
#pragma unroll
  for (int i = 0; i < 16; ++i) r[i] = (_Float16)0.0f;
  return r;
}

static __device__ __forceinline__ v8f wmma16(v16h a, v16h b, v8f c) {
  // D = A(16x32 f16) * B(32x16 f16) + C(16x16 f32)
  return __builtin_amdgcn_wmma_f32_16x16x32_f16(false, a, false, b, (short)0, c,
                                                false, false);
}

static __device__ __forceinline__ void atomAdd(float* p, float v) {
  __hip_atomic_fetch_add(p, v, __ATOMIC_RELAXED, __HIP_MEMORY_SCOPE_AGENT);
}

union H4 { uint2 u; _Float16 h[4]; };
union H8 { uint4 u; _Float16 h[8]; };

// ---------------------------------------------------------------- zero scratch
__global__ void k_zero(float* __restrict__ p, size_t n) {
  size_t i  = (size_t)blockIdx.x * blockDim.x + threadIdx.x;
  size_t st = (size_t)gridDim.x * blockDim.x;
  for (; i < n; i += st) p[i] = 0.0f;
}

// ------------------------------------------------- per-node pe MLP for conv1
// npe1[v][0..3] = mlp2(x[v], pe1)   (1 -> 4 -> 4)
__global__ void k_node_pe1(const float* __restrict__ x,
                           const float* __restrict__ w1, const float* __restrict__ b1,
                           const float* __restrict__ w2, const float* __restrict__ b2,
                           _Float16* __restrict__ npe1, int n) {
  int v = blockIdx.x * blockDim.x + threadIdx.x;
  if (v >= n) return;
  float t = x[v];
  float u[4];
#pragma unroll
  for (int j = 0; j < 4; ++j) u[j] = fmaxf(fmaf(t, w1[j], b1[j]), 0.0f);
  H4 o;
#pragma unroll
  for (int k = 0; k < 4; ++k) {
    float s = b2[k];
#pragma unroll
    for (int j = 0; j < 4; ++j) s = fmaf(u[j], w2[j * 4 + k], s);
    o.h[k] = (_Float16)s;
  }
  *(uint2*)(npe1 + 4 * (size_t)v) = o.u;
}

// ------------------------------------------------- per-node pe MLP for conv2
// npe2[v][0..7] = mlp2(relu(h1[v]), pe2)   (8 -> 8 -> 8)
__global__ void k_node_pe2(const float* __restrict__ h1,
                           const float* __restrict__ w1, const float* __restrict__ b1,
                           const float* __restrict__ w2, const float* __restrict__ b2,
                           _Float16* __restrict__ npe2, int n) {
  int v = blockIdx.x * blockDim.x + threadIdx.x;
  if (v >= n) return;
  float hin[8];
#pragma unroll
  for (int j = 0; j < 8; ++j) hin[j] = fmaxf(h1[(size_t)v * 8 + j], 0.0f);
  float u[8];
#pragma unroll
  for (int j = 0; j < 8; ++j) {
    float s = b1[j];
#pragma unroll
    for (int k = 0; k < 8; ++k) s = fmaf(hin[k], w1[k * 8 + j], s);
    u[j] = fmaxf(s, 0.0f);
  }
  H8 o;
#pragma unroll
  for (int k = 0; k < 8; ++k) {
    float s = b2[k];
#pragma unroll
    for (int j = 0; j < 8; ++j) s = fmaf(u[j], w2[j * 8 + k], s);
    o.h[k] = (_Float16)s;
  }
  *(uint4*)(npe2 + 8 * (size_t)v) = o.u;
}

// ---------------------------------------------------------------- conv1 edges
// feat(16) = [pe_i(4) | pe_j(4) | de(8)] ; msg = relu(feat*W1+b1)*W2+b2 (8)
// One wave = 16 edges = one WMMA M-tile. A-layout (16-bit 16x32):
//   lane<16 : halfs0-7 = K0-7  (pe_i,pe_j gathers), halfs8-15 = K16-23 (=0)
//   lane>=16: halfs0-7 = K8-15 (de MLP in VALU),    halfs8-15 = K24-31 (=0)
__global__ __launch_bounds__(TPB) void k_conv1(
    const int* __restrict__ SRC, const int* __restrict__ DST,
    const float2* __restrict__ pos, const _Float16* __restrict__ npe1,
    const float* __restrict__ de_w1, const float* __restrict__ de_b1,
    const float* __restrict__ de_w2, const float* __restrict__ de_b2,
    const float* __restrict__ mw1, const float* __restrict__ mb1,
    const float* __restrict__ mw2, const float* __restrict__ mb2,
    float* __restrict__ out, int ntiles) {
  __shared__ _Float16 mid[WPB][16][16] __attribute__((aligned(16)));
  const int  lane = threadIdx.x & 31;
  const int  wid  = threadIdx.x >> 5;
  const int  row  = lane & 15;
  const bool lo   = lane < 16;
  const int  col  = row;

  float dW1[8], dB1[8], dB2[8], dW2[64];
#pragma unroll
  for (int i = 0; i < 8; ++i) { dW1[i] = de_w1[i]; dB1[i] = de_b1[i]; dB2[i] = de_b2[i]; }
#pragma unroll
  for (int i = 0; i < 64; ++i) dW2[i] = de_w2[i];

  // B frag layer1: K=feat row (16 valid, pad 32) x N=8 (pad 16).
  // lane<16 halfs0-7 hold K=0..7, lane>=16 halfs0-7 hold K=8..15 for column `col`; K16-31 zero.
  v16h bf1 = zv16();
  if (col < 8) {
#pragma unroll
    for (int i = 0; i < 8; ++i) bf1[i] = (_Float16)mw1[((lo ? 0 : 8) + i) * 8 + col];
  }
  // B frag layer2: W2 is 8x8.
  v16h bf2 = zv16();
  if (lo && col < 8) {
#pragma unroll
    for (int i = 0; i < 8; ++i) bf2[i] = (_Float16)mw2[i * 8 + col];
  }
  v8f c1, c2;
#pragma unroll
  for (int r = 0; r < 8; ++r) {
    c1[r] = (col < 8) ? mb1[col] : 0.0f;
    c2[r] = (col < 8) ? mb2[col] : 0.0f;
  }

  const int wavesTotal = gridDim.x * WPB;
  for (int t = blockIdx.x * WPB + wid; t < ntiles; t += wavesTotal) {
    const int e  = t * 16 + row;
    const int sv = SRC[e];
    const int dv = DST[e];
    const int dOff = dv * 8;   // 32-bit scaled scatter base, shuffled below
    v16h a = zv16();
    if (lo) {
      H4 pi, pj;
      pi.u = *(const uint2*)(npe1 + 4 * (size_t)dv);
      pj.u = *(const uint2*)(npe1 + 4 * (size_t)sv);
#pragma unroll
      for (int i = 0; i < 4; ++i) { a[i] = pi.h[i]; a[4 + i] = pj.h[i]; }
    } else {
      float2 ps = pos[sv], pd = pos[dv];
      float dx = pd.x - ps.x, dy = pd.y - ps.y;
      float dist = sqrtf(dx * dx + dy * dy);
      float u[8];
#pragma unroll
      for (int j = 0; j < 8; ++j) u[j] = fmaxf(fmaf(dist, dW1[j], dB1[j]), 0.0f);
#pragma unroll
      for (int k = 0; k < 8; ++k) {
        float s = dB2[k];
#pragma unroll
        for (int j = 0; j < 8; ++j) s = fmaf(u[j], dW2[j * 8 + k], s);
        a[k] = (_Float16)s;
      }
    }
    v8f acc = wmma16(a, bf1, c1);
    // relu -> per-wave LDS tile [edge16][feat16]; C-layout: lane=col, vgpr r=row
#pragma unroll
    for (int r = 0; r < 8; ++r)
      mid[wid][r + (lo ? 0 : 8)][col] = (_Float16)fmaxf(acc[r], 0.0f);

    const int kb = lo ? 0 : 8;
    H8 m;
    m.u = *(const uint4*)&mid[wid][row][kb];
    v16h a2 = zv16();
#pragma unroll
    for (int i = 0; i < 8; ++i) a2[i] = m.h[i];

    v8f o = wmma16(a2, bf2, c2);

    // scatter: lane = output col, vgpr r = edge row (M or M+8)
    unsigned soff[8];
#pragma unroll
    for (int r = 0; r < 8; ++r) {
      int M   = r + (lo ? 0 : 8);
      soff[r] = (unsigned)(__shfl(dOff, M, 32) + col);
    }
    if (col < 8) {
#pragma unroll
      for (int r = 0; r < 8; ++r) atomAdd(out + soff[r], o[r]);
    }
  }
}

// ---------------------------------------------------------------- conv2 edges
// Our feature order: [pe_i(8) | de(8) | pe_j(8)] (W1 rows permuted to match;
// reference order is [pe_i | pe_j | de]).
//   lane<16 : halfs0-7=K0-7 (npe2[dst]), halfs8-15=K16-23 (npe2[src])
//   lane>=16: halfs0-7=K8-15 (de MLP),   halfs8-15=K24-31 (=0)
__global__ __launch_bounds__(TPB) void k_conv2(
    const int* __restrict__ SRC, const int* __restrict__ DST,
    const float2* __restrict__ pos, const _Float16* __restrict__ npe2,
    const float* __restrict__ de_w1, const float* __restrict__ de_b1,
    const float* __restrict__ de_w2, const float* __restrict__ de_b2,
    const float* __restrict__ mw1, const float* __restrict__ mb1,
    const float* __restrict__ mw2, const float* __restrict__ mb2,
    float* __restrict__ out, int ntiles) {
  __shared__ _Float16 mid[WPB][16][16] __attribute__((aligned(16)));
  const int  lane = threadIdx.x & 31;
  const int  wid  = threadIdx.x >> 5;
  const int  row  = lane & 15;
  const bool lo   = lane < 16;
  const int  col  = row;

  float dW1[8], dB1[8], dB2[8], dW2[64];
#pragma unroll
  for (int i = 0; i < 8; ++i) { dW1[i] = de_w1[i]; dB1[i] = de_b1[i]; dB2[i] = de_b2[i]; }
#pragma unroll
  for (int i = 0; i < 64; ++i) dW2[i] = de_w2[i];

  // B frag layer1 (W1 is [24][16] row-major, reference row order):
  //  ourK 0-7 -> ref rows 0-7 (pe_i); ourK 8-15 -> ref rows 16-23 (de);
  //  ourK 16-23 -> ref rows 8-15 (pe_j); ourK 24-31 -> 0
  v16h bf1 = zv16();
  if (lo) {
#pragma unroll
    for (int i = 0; i < 8; ++i)  bf1[i] = (_Float16)mw1[i * 16 + col];
#pragma unroll
    for (int i = 8; i < 16; ++i) bf1[i] = (_Float16)mw1[i * 16 + col];
  } else {
#pragma unroll
    for (int i = 0; i < 8; ++i)  bf1[i] = (_Float16)mw1[(16 + i) * 16 + col];
  }
  // B frag layer2: W2 is 16x16
  v16h bf2 = zv16();
  {
#pragma unroll
    for (int i = 0; i < 8; ++i) bf2[i] = (_Float16)mw2[((lo ? 0 : 8) + i) * 16 + col];
  }
  v8f c1, c2;
#pragma unroll
  for (int r = 0; r < 8; ++r) { c1[r] = mb1[col]; c2[r] = mb2[col]; }

  const int wavesTotal = gridDim.x * WPB;
  for (int t = blockIdx.x * WPB + wid; t < ntiles; t += wavesTotal) {
    const int e  = t * 16 + row;
    const int sv = SRC[e];
    const int dv = DST[e];
    const int dOff = dv * 16;
    v16h a = zv16();
    if (lo) {
      H8 pi, pj;
      pi.u = *(const uint4*)(npe2 + 8 * (size_t)dv);
      pj.u = *(const uint4*)(npe2 + 8 * (size_t)sv);
#pragma unroll
      for (int i = 0; i < 8; ++i) { a[i] = pi.h[i]; a[8 + i] = pj.h[i]; }
    } else {
      float2 ps = pos[sv], pd = pos[dv];
      float dx = pd.x - ps.x, dy = pd.y - ps.y;
      float dist = sqrtf(dx * dx + dy * dy);
      float u[8];
#pragma unroll
      for (int j = 0; j < 8; ++j) u[j] = fmaxf(fmaf(dist, dW1[j], dB1[j]), 0.0f);
#pragma unroll
      for (int k = 0; k < 8; ++k) {
        float s = dB2[k];
#pragma unroll
        for (int j = 0; j < 8; ++j) s = fmaf(u[j], dW2[j * 8 + k], s);
        a[k] = (_Float16)s;
      }
    }
    v8f acc = wmma16(a, bf1, c1);
#pragma unroll
    for (int r = 0; r < 8; ++r)
      mid[wid][r + (lo ? 0 : 8)][col] = (_Float16)fmaxf(acc[r], 0.0f);

    const int kb = lo ? 0 : 8;
    H8 m;
    m.u = *(const uint4*)&mid[wid][row][kb];
    v16h a2 = zv16();
#pragma unroll
    for (int i = 0; i < 8; ++i) a2[i] = m.h[i];

    v8f o = wmma16(a2, bf2, c2);

    unsigned soff[8];
#pragma unroll
    for (int r = 0; r < 8; ++r) {
      int M   = r + (lo ? 0 : 8);
      soff[r] = (unsigned)(__shfl(dOff, M, 32) + col);
    }
#pragma unroll
    for (int r = 0; r < 8; ++r) atomAdd(out + soff[r], o[r]);
  }
}

// ------------------------------------------------- relu + global_add_pool
__global__ void k_pool(const float* __restrict__ h2, const int* __restrict__ batch,
                       float* __restrict__ g, int n) {
  int i = blockIdx.x * blockDim.x + threadIdx.x;
  if (i >= n * 16) return;
  int v = i >> 4, c = i & 15;
  float val = fmaxf(h2[i], 0.0f);
  atomAdd(g + (size_t)batch[v] * 16 + c, val);
}

// ------------------------------------------------- readout MLP (128 graphs)
__global__ void k_readout(const float* __restrict__ g,
                          const float* __restrict__ w1,
                          const float* __restrict__ gamma, const float* __restrict__ beta,
                          const float* __restrict__ w2, const float* __restrict__ b2,
                          float* __restrict__ out) {
  int i = threadIdx.x;
  if (i >= 128) return;
  float gi[16];
#pragma unroll
  for (int j = 0; j < 16; ++j) gi[j] = g[i * 16 + j];
  float invs = rsqrtf(1.0f + 1e-5f);
  float z[8];
#pragma unroll
  for (int k = 0; k < 8; ++k) {
    float s = 0.0f;
#pragma unroll
    for (int j = 0; j < 16; ++j) s = fmaf(gi[j], w1[j * 8 + k], s);
    z[k] = fmaxf(fmaf(s * invs, gamma[k], beta[k]), 0.0f);
  }
#pragma unroll
  for (int k = 0; k < 8; ++k) {
    float s = b2[k];
#pragma unroll
    for (int j = 0; j < 8; ++j) s = fmaf(z[j], w2[j * 8 + k], s);
    out[i * 8 + k] = s;
  }
}

// ----------------------------------------------------------------- launcher
extern "C" void kernel_launch(void* const* d_in, const int* in_sizes, int n_in,
                              void* d_out, int out_size, void* d_ws, size_t ws_size,
                              hipStream_t stream) {
  const float* x    = (const float*)d_in[0];
  const float* pos  = (const float*)d_in[1];
  const int*   ei   = (const int*)d_in[2];
  const int*   batc = (const int*)d_in[3];
  const float* P[29];
  for (int i = 0; i < 29; ++i) P[i] = (const float*)d_in[4 + i];
  // P: 0-3 c1_pe_{w1,b1,w2,b2}  4-7 c1_de_*  8-11 c1_mlp_*
  //    12-15 c2_pe_*  16-19 c2_de_*  20-23 c2_mlp_*
  //    24 ro_w1  25 gamma  26 beta  27 ro_w2  28 ro_b2

  const int n      = in_sizes[0];
  const int E      = in_sizes[2] / 2;
  const int ntiles = E / 16;   // E = 6.4M -> 400000 exact tiles

  auto al = [](size_t v) { return (v + 255) & ~(size_t)255; };
  char* ws = (char*)d_ws;
  size_t off = 0;
  float*    h1   = (float*)(ws + off);    off += al((size_t)n * 8 * sizeof(float));
  float*    h2   = (float*)(ws + off);    off += al((size_t)n * 16 * sizeof(float));
  _Float16* npe1 = (_Float16*)(ws + off); off += al((size_t)n * 4 * sizeof(_Float16));
  _Float16* npe2 = (_Float16*)(ws + off); off += al((size_t)n * 8 * sizeof(_Float16));
  float*    g    = (float*)(ws + off);    off += al((size_t)128 * 16 * sizeof(float));

  k_zero<<<2048, 256, 0, stream>>>((float*)d_ws, off / sizeof(float));

  k_node_pe1<<<(n + 255) / 256, 256, 0, stream>>>(x, P[0], P[1], P[2], P[3], npe1, n);

  const int convBlocks = 3200;  // 25600 waves, grid-stride over 400k tiles
  k_conv1<<<convBlocks, TPB, 0, stream>>>(ei, ei + E, (const float2*)pos, npe1,
                                          P[4], P[5], P[6], P[7],
                                          P[8], P[9], P[10], P[11], h1, ntiles);

  k_node_pe2<<<(n + 255) / 256, 256, 0, stream>>>(h1, P[12], P[13], P[14], P[15], npe2, n);

  k_conv2<<<convBlocks, TPB, 0, stream>>>(ei, ei + E, (const float2*)pos, npe2,
                                          P[16], P[17], P[18], P[19],
                                          P[20], P[21], P[22], P[23], h2, ntiles);

  k_pool<<<((n * 16) + 255) / 256, 256, 0, stream>>>(h2, batc, g, n);

  k_readout<<<1, 128, 0, stream>>>(g, P[24], P[25], P[26], P[27], P[28], (float*)d_out);
}
